// Qwen3MoeGroupedExperts_35691178230103
// MI455X (gfx1250) — matrix operand order repaired and checked by reading, not compiled, not run on hardware
//
#include <hip/hip_runtime.h>

// ---------------------------------------------------------------------------
// Qwen3 MoE grouped experts (B=4,S=1024,H=1024,I=768,E=8,K=2), fp32 in/out.
// Strategy: expert-dispatch (each token only visits its K=2 experts),
// bf16 WMMA (v_wmma_f32_16x16x32_bf16) for all three GEMMs, f32 accum,
// fused SiLU, atomic-add combine.  Workspace use: ~36.6 MiB
// (dispatch lists 0.5 MiB + bf16 weight copies 3x12 MiB).
// ---------------------------------------------------------------------------

#define T_TOK   4096
#define HDIM    1024
#define IDIM    768
#define NEXP    8
#define TOPK    2
#define CAP     (T_TOK * TOPK)          // worst-case tokens per expert
#define TILE_M  32                      // tokens per block tile
#define XSTRIDE (HDIM + 8)              // +16B pad -> conflict-free A frags
#define ASTRIDE (IDIM + 8)

typedef __attribute__((ext_vector_type(16))) __bf16          v16bf;
typedef __attribute__((ext_vector_type(8)))  float           v8f;
typedef __attribute__((ext_vector_type(8)))  unsigned short  v8u16;
typedef __attribute__((ext_vector_type(4)))  unsigned short  v4u16;

union BfPack { v8u16 h[2]; v16bf bf; };

static __device__ __forceinline__ unsigned short f2bf(float f) {
    unsigned int u = __float_as_uint(f);
    u = (u + 0x7FFFu + ((u >> 16) & 1u)) >> 16;   // round-to-nearest-even
    return (unsigned short)u;
}

// A fragment (16x32 bf16, M x K), source row-major [m][k] in LDS.
// Lane L: row m = m0 + L%16; half = L/16.
// VGPR0-3 : K = 8*half .. 8*half+7   (16 contiguous bytes)
// VGPR4-7 : K = 16+8*half .. +7      (16 contiguous bytes)
static __device__ __forceinline__ v16bf load_a_lds(const unsigned short* X,
                                                   int stride, int m0, int k0,
                                                   int lane) {
    const unsigned short* p =
        X + (m0 + (lane & 15)) * stride + k0 + ((lane >> 4) << 3);
    BfPack pk;
    pk.h[0] = *(const v8u16*)(p);
    pk.h[1] = *(const v8u16*)(p + 16);
    return pk.bf;
}

// B fragment (32x16 bf16, K x N), weight row-major [n][k] (out-major) in global.
// Lane L: col n = n0 + L%16; K = 8*half .. +7 and 16+8*half .. +7 (16 bytes each).
static __device__ __forceinline__ v16bf load_b_glb(const unsigned short* W,
                                                   int ld, int n0, int k0,
                                                   int lane) {
    const unsigned short* p =
        W + (size_t)(n0 + (lane & 15)) * ld + k0 + ((lane >> 4) << 3);
    BfPack pk;
    pk.h[0] = *(const v8u16*)(p);
    pk.h[1] = *(const v8u16*)(p + 16);
    return pk.bf;
}

// ---------------------------------------------------------------------------
// fp32 -> bf16 weight conversion (one full pass; bf16 copies live in L2)
// ---------------------------------------------------------------------------
__global__ void moe_cvt_bf16(const float4* __restrict__ src,
                             v4u16* __restrict__ dst, int n4) {
    for (int i = blockIdx.x * blockDim.x + threadIdx.x; i < n4;
         i += gridDim.x * blockDim.x) {
        float4 v = src[i];
        v4u16 p = { f2bf(v.x), f2bf(v.y), f2bf(v.z), f2bf(v.w) };
        dst[i] = p;
    }
}

// ---------------------------------------------------------------------------
// Build per-expert token lists (counts must be zeroed before launch).
// ---------------------------------------------------------------------------
__global__ void moe_dispatch(const int* __restrict__ sel,
                             const float* __restrict__ rw,
                             int* __restrict__ counts,
                             int* __restrict__ tok,
                             float* __restrict__ wgt) {
    int t = blockIdx.x * blockDim.x + threadIdx.x;
    if (t >= T_TOK) return;
#pragma unroll
    for (int k = 0; k < TOPK; ++k) {
        int e = sel[t * TOPK + k] & (NEXP - 1);
        float wv = rw[t * TOPK + k];
        int slot = atomicAdd(&counts[e], 1);
        tok[e * CAP + slot] = t;
        wgt[e * CAP + slot] = wv;
    }
}

// ---------------------------------------------------------------------------
// Main fused kernel: one block = (expert e, 32-token tile).
//   GEMM1/2: act = silu(x Wg^T) * (x Wu^T)    (K = 1024)
//   GEMM3  : y   = act Wd^T ; out[t] += w_t*y (K = 768, atomic combine)
// 8 wave32s; each wave owns (m-tile = w&1, striped 64-wide N chunks).
// ---------------------------------------------------------------------------
__global__ __launch_bounds__(256) void moe_expert_kernel(
    const float* __restrict__ hidden, const int* __restrict__ counts,
    const int* __restrict__ tok, const float* __restrict__ wgt,
    const unsigned short* __restrict__ Wg, const unsigned short* __restrict__ Wu,
    const unsigned short* __restrict__ Wd, float* __restrict__ out) {
    const int e = blockIdx.y;
    const int cnt = counts[e];
    const int tile0 = blockIdx.x * TILE_M;
    if (tile0 >= cnt) return;

    extern __shared__ char smem_raw[];
    int* tile_tok = (int*)smem_raw;                                 // 128 B
    float* tile_w = (float*)(smem_raw + 256);                       // 128 B
    unsigned short* x_lds = (unsigned short*)(smem_raw + 512);      // 32x1032 bf16
    unsigned short* act_lds =
        (unsigned short*)(smem_raw + 512 + TILE_M * XSTRIDE * 2);   // 32x776 bf16

    const int tid = threadIdx.x;
    if (tid < TILE_M) {
        int entry = tile0 + tid;
        if (entry < cnt) {
            tile_tok[tid] = tok[e * CAP + entry];
            tile_w[tid] = wgt[e * CAP + entry];
        } else {
            tile_tok[tid] = 0;
            tile_w[tid] = 0.0f;   // padded rows contribute exactly zero
        }
    }
    __syncthreads();

    // Gather 32 token rows of x -> bf16 in LDS (float4 reads, 8B LDS stores).
    const float4* h4 = (const float4*)hidden;
    for (int idx = tid; idx < TILE_M * (HDIM / 4); idx += 256) {
        int r = idx >> 8;            // HDIM/4 = 256 chunks per row
        int c = idx & 255;
        int t = tile_tok[r];
        float4 v = h4[(size_t)t * (HDIM / 4) + c];
        v4u16 p = { f2bf(v.x), f2bf(v.y), f2bf(v.z), f2bf(v.w) };
        *(v4u16*)(x_lds + r * XSTRIDE + (c << 2)) = p;
    }
    __syncthreads();

    const int w = tid >> 5;
    const int lane = tid & 31;
    const int m0 = (w & 1) << 4;
    const int half = lane >> 4;
    const int nn = lane & 15;
    const size_t eoff = (size_t)e * IDIM * HDIM;
    const unsigned short* wg_e = Wg + eoff;
    const unsigned short* wu_e = Wu + eoff;
    const unsigned short* wd_e = Wd + eoff;   // [H][I], same element count

    const v8f vzero = {0.f, 0.f, 0.f, 0.f, 0.f, 0.f, 0.f, 0.f};

    // ---- GEMM 1+2 (gate & up), fused SiLU, act -> LDS ----
    for (int i = 0; i < 3; ++i) {
        int n0 = ((w >> 1) + (i << 2)) << 6;   // 64-wide N chunk, 12 chunks total
        v8f g[4], u[4];
#pragma unroll
        for (int j = 0; j < 4; ++j) { g[j] = vzero; u[j] = vzero; }
        for (int k = 0; k < HDIM; k += 32) {
            v16bf a = load_a_lds(x_lds, XSTRIDE, m0, k, lane);
#pragma unroll
            for (int j = 0; j < 4; ++j) {
                v16bf bg = load_b_glb(wg_e, HDIM, n0 + (j << 4), k, lane);
                g[j] = __builtin_amdgcn_wmma_f32_16x16x32_bf16(
                    false, a, false, bg, (short)0, g[j], false, false);
                v16bf bu = load_b_glb(wu_e, HDIM, n0 + (j << 4), k, lane);
                u[j] = __builtin_amdgcn_wmma_f32_16x16x32_bf16(
                    false, a, false, bu, (short)0, u[j], false, false);
            }
        }
#pragma unroll
        for (int j = 0; j < 4; ++j) {
            int n = n0 + (j << 4) + nn;
#pragma unroll
            for (int r = 0; r < 8; ++r) {
                int m = m0 + r + (half << 3);
                float gg = g[j][r];
                float s = gg / (1.0f + __expf(-gg));   // SiLU
                act_lds[m * ASTRIDE + n] = f2bf(s * u[j][r]);
            }
        }
    }
    __syncthreads();

    // ---- GEMM 3 (down) + weighted atomic combine ----
    for (int i = 0; i < 4; ++i) {
        int h0 = ((w >> 1) + (i << 2)) << 6;   // 64-wide H chunk, 16 chunks total
        v8f y[4];
#pragma unroll
        for (int j = 0; j < 4; ++j) y[j] = vzero;
        for (int k = 0; k < IDIM; k += 32) {
            v16bf a = load_a_lds(act_lds, ASTRIDE, m0, k, lane);
#pragma unroll
            for (int j = 0; j < 4; ++j) {
                v16bf bd = load_b_glb(wd_e, IDIM, h0 + (j << 4), k, lane);
                y[j] = __builtin_amdgcn_wmma_f32_16x16x32_bf16(
                    false, a, false, bd, (short)0, y[j], false, false);
            }
        }
#pragma unroll
        for (int j = 0; j < 4; ++j) {
#pragma unroll
            for (int r = 0; r < 8; ++r) {
                int m = m0 + r + (half << 3);
                int t = tile_tok[m];
                float ww = tile_w[m];
                unsafeAtomicAdd(&out[(size_t)t * HDIM + h0 + (j << 4) + nn],
                                y[j][r] * ww);
            }
        }
    }
}

// ---------------------------------------------------------------------------
extern "C" void kernel_launch(void* const* d_in, const int* in_sizes, int n_in,
                              void* d_out, int out_size, void* d_ws,
                              size_t ws_size, hipStream_t stream) {
    const float* hidden  = (const float*)d_in[0];
    const float* routing = (const float*)d_in[1];
    const int*   sel     = (const int*)d_in[2];
    const float* gateW   = (const float*)d_in[3];
    const float* upW     = (const float*)d_in[4];
    const float* downW   = (const float*)d_in[5];
    float* out = (float*)d_out;

    // Workspace layout (needs ~36.6 MiB):
    char* ws = (char*)d_ws;
    int*   counts = (int*)ws;                                   // 256 B
    int*   tok    = (int*)(ws + 256);                           // CAP*NEXP ints
    float* wgt    = (float*)(ws + 256 + NEXP * CAP * 4);        // CAP*NEXP f32
    const size_t wbytes = (size_t)NEXP * IDIM * HDIM * 2;       // 12 MiB each
    unsigned short* wgB = (unsigned short*)(ws + 256 + 2 * NEXP * CAP * 4);
    unsigned short* wuB = wgB + (size_t)NEXP * IDIM * HDIM;
    unsigned short* wdB = wuB + (size_t)NEXP * IDIM * HDIM;
    (void)wbytes; (void)in_sizes; (void)n_in; (void)out_size; (void)ws_size;

    hipMemsetAsync(counts, 0, 256, stream);
    hipMemsetAsync(out, 0, (size_t)T_TOK * HDIM * sizeof(float), stream);

    const int n4 = NEXP * IDIM * HDIM / 4;
    moe_cvt_bf16<<<2048, 256, 0, stream>>>((const float4*)gateW, (v4u16*)wgB, n4);
    moe_cvt_bf16<<<2048, 256, 0, stream>>>((const float4*)upW,   (v4u16*)wuB, n4);
    moe_cvt_bf16<<<2048, 256, 0, stream>>>((const float4*)downW, (v4u16*)wdB, n4);

    moe_dispatch<<<T_TOK / 256, 256, 0, stream>>>(sel, routing, counts, tok, wgt);

    const size_t lds_bytes =
        512 + (size_t)TILE_M * XSTRIDE * 2 + (size_t)TILE_M * ASTRIDE * 2; // 116,224 B
    dim3 grid(CAP / TILE_M, NEXP);   // worst-case tiles; blocks early-exit
    moe_expert_kernel<<<grid, 256, lds_bytes, stream>>>(
        hidden, counts, tok, wgt, wgB, wuB, wdB, out);
}
